// ESLayer_2886218023208
// MI455X (gfx1250) — compile-verified, matrix operand order checked by reading, not run on hardware
//
#include <hip/hip_runtime.h>

// Problem constants (match reference)
#define NN  20000     // nodes
#define NE  320000    // edges
#define TT  4         // time steps
#define HHC 64        // node hidden
#define HEC 16        // edge attr
#define HDC 128       // MLP hidden
#define K1P 160       // edge-MLP K (145) padded to multiple of 32
#define KNP 192       // node-MLP K (64+128)

typedef __attribute__((ext_vector_type(16))) __bf16 v16bf;
typedef __attribute__((ext_vector_type(8)))  __bf16 v8bf;
typedef __attribute__((ext_vector_type(8)))  float  v8f;

__device__ __forceinline__ __bf16 f2bf(float f) {
  unsigned u = __builtin_bit_cast(unsigned, f);
  unsigned r = u + 0x7fffu + ((u >> 16) & 1u);        // round-to-nearest-even
  unsigned short hs = (unsigned short)(r >> 16);
  return __builtin_bit_cast(__bf16, hs);
}
__device__ __forceinline__ float bf2f(__bf16 b) {
  unsigned short hs = __builtin_bit_cast(unsigned short, b);
  unsigned u = ((unsigned)hs) << 16;
  return __builtin_bit_cast(float, u);
}
__device__ __forceinline__ float silu(float v) { return v / (1.0f + __expf(-v)); }

// Assemble a 16x32 bf16 WMMA fragment from a row-major [16 x stride] block.
// ISA layout (CDNA5 ISA): lane&15 = M row; lane-half selects K {0..7,16..23}
// vs {8..15,24..31}. Two 16B loads per fragment.
__device__ __forceinline__ v16bf load_frag(const __bf16* p, int stride, int lane, int kbase) {
  const int m  = lane & 15;
  const int ko = kbase + ((lane >> 4) << 3);
  const __bf16* q = p + m * stride + ko;
  union { v16bf v; v8bf h[2]; } u;
  u.h[0] = *(const v8bf*)(q);
  u.h[1] = *(const v8bf*)(q + 16);
  return u.v;
}

// One wave computes a 16-row stripe x (NNT*16) cols; A from LDS, B (=W^T, [n][k]) from global.
template <int NKT, int NNT>
__device__ __forceinline__ void gemm_bf16(const __bf16* Ablk, int lda,
                                          const __bf16* B, int ldb,
                                          int lane, v8f* acc) {
#pragma unroll
  for (int kt = 0; kt < NKT; ++kt) {
    v16bf a = load_frag(Ablk, lda, lane, kt * 32);
#pragma unroll
    for (int nt = 0; nt < NNT; ++nt) {
      v16bf b = load_frag(B + nt * 16 * ldb, ldb, lane, kt * 32);
      acc[nt] = __builtin_amdgcn_wmma_f32_16x16x32_bf16(
          false, a, false, b, (short)0, acc[nt], false, false);
    }
  }
}

// ---------------------------------------------------------------------------
// Zero the accumulation region of the workspace (must run every launch).
__global__ void egnn_zero_kernel(float* __restrict__ p, int n) {
  for (int i = blockIdx.x * blockDim.x + threadIdx.x; i < n; i += gridDim.x * blockDim.x)
    p[i] = 0.0f;
}

// Pre-transpose all weights to bf16, layout Wt[n][k] (B-fragment friendly).
__global__ void egnn_convert_weights(
    const float* __restrict__ ew1, const float* __restrict__ ew2,
    const float* __restrict__ cw1, const float* __restrict__ nw1,
    const float* __restrict__ nw2,
    __bf16* __restrict__ W1e, __bf16* __restrict__ W2e, __bf16* __restrict__ Wc1,
    __bf16* __restrict__ W1n, __bf16* __restrict__ W2n)
{
  const int S1 = HDC * K1P, S2 = HDC * HDC, S3 = HDC * HDC, S4 = HDC * KNP, S5 = HHC * HDC;
  int i = blockIdx.x * blockDim.x + threadIdx.x;
  if (i < S1) {
    int n = i / K1P, k = i - n * K1P;
    W1e[i] = f2bf(k < 145 ? ew1[k * HDC + n] : 0.0f);
  } else if (i < S1 + S2) {
    int j = i - S1; int n = j / HDC, k = j - n * HDC;
    W2e[j] = f2bf(ew2[k * HDC + n]);
  } else if (i < S1 + S2 + S3) {
    int j = i - S1 - S2; int n = j / HDC, k = j - n * HDC;
    Wc1[j] = f2bf(cw1[k * HDC + n]);
  } else if (i < S1 + S2 + S3 + S4) {
    int j = i - S1 - S2 - S3; int n = j / KNP, k = j - n * KNP;
    W1n[j] = f2bf(nw1[k * HDC + n]);
  } else if (i < S1 + S2 + S3 + S4 + S5) {
    int j = i - S1 - S2 - S3 - S4; int n = j / HDC, k = j - n * HDC;
    W2n[j] = f2bf(nw2[k * HHC + n]);
  }
}

// ---------------------------------------------------------------------------
// Fused edge kernel: 16 edges x T=4 -> 64 GEMM rows per workgroup.
// edge-MLP(2 layers, SiLU) -> m_ij ; atomic segment-sum into m_i ;
// coord-MLP(layer1 WMMA, layer2 dot) -> weighted x_ij scatter into agg.
__global__ __launch_bounds__(128, 1) void egnn_edge_kernel(
    const float* __restrict__ x, const float* __restrict__ h,
    const int* __restrict__ eidx, const float* __restrict__ eattr,
    const __bf16* __restrict__ W1, const float* __restrict__ b1,
    const __bf16* __restrict__ W2, const float* __restrict__ b2,
    const __bf16* __restrict__ Wc1, const float* __restrict__ bc1,
    const float* __restrict__ wc2, const float* __restrict__ bc2,
    float* __restrict__ m_i, float* __restrict__ agg, float* __restrict__ cnt)
{
  __shared__ alignas(16) __bf16 Xin[64 * K1P];   // edge-MLP input (reused as C1 later)
  __shared__ alignas(16) __bf16 Act[64 * HDC];   // hidden activations
  __shared__ alignas(16) __bf16 Mij[64 * HDC];   // m_ij (bf16 copy for coord MLP)
  __shared__ int   srow[16];
  __shared__ int   scol[16];
  __shared__ float xij[64 * 3];
  __shared__ float nrm[64];

  const int tid  = threadIdx.x;
  const int lane = tid & 31;
  const int wid  = tid >> 5;
  const int e0   = blockIdx.x * 16;

  if (tid < 16) {
    srow[tid] = eidx[e0 + tid];
    scol[tid] = eidx[NE + e0 + tid];
  }
  __syncthreads();
  if (tid < 16) atomicAdd(&cnt[srow[tid]], 1.0f);   // one per edge

  // x_ij and squared norm per (edge,t) row
  if (tid < 64) {
    const int le = tid >> 2, t = tid & 3;
    const int na = srow[le], nb = scol[le];
    float s = 0.0f;
#pragma unroll
    for (int d = 0; d < 3; ++d) {
      float v = x[(na * 3 + d) * TT + t] - x[(nb * 3 + d) * TT + t];
      xij[tid * 3 + d] = v;
      s += v * v;
    }
    nrm[tid] = s;
    Xin[tid * K1P + 2 * HHC] = f2bf(s);            // k == 128
  }
  // gather [h_row | h_col | norm2 | edge_attr | 0-pad] into LDS (bf16)
  for (int idx = tid; idx < 64 * K1P; idx += 128) {
    const int r = idx / K1P, k = idx - r * K1P;
    const int le = r >> 2, t = r & 3;
    float v;
    if (k < HHC)                     v = h[(srow[le] * HHC + k) * TT + t];
    else if (k < 2 * HHC)            v = h[(scol[le] * HHC + (k - HHC)) * TT + t];
    else if (k == 2 * HHC)           continue;     // written above
    else if (k < 2 * HHC + 1 + HEC)  v = eattr[((e0 + le) * HEC + (k - 129)) * TT + t];
    else                             v = 0.0f;
    Xin[idx] = f2bf(v);
  }
  __syncthreads();

  const int colbase = lane & 15;
  const int rsub    = (lane >> 4) << 3;            // D-matrix row offset: 0 or 8

  // ---- edge MLP layer 1: [16x160] x [160x128], +b1, SiLU -> Act ----
  {
    v8f acc[8] = {};
    gemm_bf16<5, 8>(Xin + wid * 16 * K1P, K1P, W1, K1P, lane, acc);
#pragma unroll
    for (int nt = 0; nt < 8; ++nt) {
      const int col = colbase + nt * 16;
      const float bb = b1[col];
#pragma unroll
      for (int i = 0; i < 8; ++i) {
        const int rl = wid * 16 + rsub + i;
        Act[rl * HDC + col] = f2bf(silu(acc[nt][i] + bb));
      }
    }
  }
  // ---- edge MLP layer 2: [16x128] x [128x128], +b2, SiLU -> m_ij ----
  {
    v8f acc[8] = {};
    gemm_bf16<4, 8>(Act + wid * 16 * HDC, HDC, W2, HDC, lane, acc);
#pragma unroll
    for (int nt = 0; nt < 8; ++nt) {
      const int col = colbase + nt * 16;
      const float bb = b2[col];
#pragma unroll
      for (int i = 0; i < 8; ++i) {
        const int rl = wid * 16 + rsub + i;
        const float v = silu(acc[nt][i] + bb);
        Mij[rl * HDC + col] = f2bf(v);
        const int le = rl >> 2, t = rl & 3;
        atomicAdd(&m_i[(srow[le] * HDC + col) * TT + t], v);  // segment_sum
      }
    }
  }
  __syncthreads();   // all waves done reading Xin before reuse as C1

  // ---- coord MLP layer 1: m_ij x [128x128], +bc1, SiLU -> C1 (aliases Xin) ----
  __bf16* C1 = Xin;
  {
    v8f acc[8] = {};
    gemm_bf16<4, 8>(Mij + wid * 16 * HDC, HDC, Wc1, HDC, lane, acc);
#pragma unroll
    for (int nt = 0; nt < 8; ++nt) {
      const int col = colbase + nt * 16;
      const float bb = bc1[col];
#pragma unroll
      for (int i = 0; i < 8; ++i) {
        const int rl = wid * 16 + rsub + i;
        C1[rl * HDC + col] = f2bf(silu(acc[nt][i] + bb));
      }
    }
  }
  __syncthreads();

  // ---- coord MLP layer 2 (128->1 dot), normalize, scatter q*x_ij ----
  if (tid < 64) {
    float s = bc2[0];
    for (int c = 0; c < HDC; ++c) s += bf2f(C1[tid * HDC + c]) * wc2[c];
    const float q = s / (sqrtf(nrm[tid] + 1e-8f) + 1.0f);
    const int le = tid >> 2, t = tid & 3;
    const int node = srow[le];
#pragma unroll
    for (int d = 0; d < 3; ++d)
      atomicAdd(&agg[(node * 3 + d) * TT + t], q * xij[tid * 3 + d]);
  }
}

// ---------------------------------------------------------------------------
// Node kernel: 16 nodes x T=4 -> 64 rows. [h|m_i] -> MLP(2 layers) -> h_new.
__global__ __launch_bounds__(128, 1) void egnn_node_kernel(
    const float* __restrict__ h, const float* __restrict__ m_i,
    const __bf16* __restrict__ W1, const float* __restrict__ b1,
    const __bf16* __restrict__ W2, const float* __restrict__ b2,
    float* __restrict__ hout)
{
  __shared__ alignas(16) __bf16 Xn[64 * KNP];
  __shared__ alignas(16) __bf16 Act[64 * HDC];
  const int tid = threadIdx.x, lane = tid & 31, wid = tid >> 5;
  const int n0 = blockIdx.x * 16;

  for (int idx = tid; idx < 64 * KNP; idx += 128) {
    const int r = idx / KNP, k = idx - r * KNP;
    const int node = n0 + (r >> 2), t = r & 3;
    const float v = (k < HHC) ? h[(node * HHC + k) * TT + t]
                              : m_i[(node * HDC + (k - HHC)) * TT + t];
    Xn[idx] = f2bf(v);
  }
  __syncthreads();

  const int colbase = lane & 15;
  const int rsub    = (lane >> 4) << 3;

  {
    v8f acc[8] = {};
    gemm_bf16<6, 8>(Xn + wid * 16 * KNP, KNP, W1, KNP, lane, acc);
#pragma unroll
    for (int nt = 0; nt < 8; ++nt) {
      const int col = colbase + nt * 16;
      const float bb = b1[col];
#pragma unroll
      for (int i = 0; i < 8; ++i) {
        const int rl = wid * 16 + rsub + i;
        Act[rl * HDC + col] = f2bf(silu(acc[nt][i] + bb));
      }
    }
  }
  {
    v8f acc[4] = {};
    gemm_bf16<4, 4>(Act + wid * 16 * HDC, HDC, W2, HDC, lane, acc);
#pragma unroll
    for (int nt = 0; nt < 4; ++nt) {
      const int col = colbase + nt * 16;      // 0..63
      const float bb = b2[col];
#pragma unroll
      for (int i = 0; i < 8; ++i) {
        const int rl = wid * 16 + rsub + i;
        const int node = n0 + (rl >> 2), t = rl & 3;
        hout[(node * HHC + col) * TT + t] = acc[nt][i] + bb;   // no final act
      }
    }
  }
}

// ---------------------------------------------------------------------------
__global__ void egnn_x_final(const float* __restrict__ x, const float* __restrict__ agg,
                             const float* __restrict__ cnt, float* __restrict__ xout) {
  int i = blockIdx.x * blockDim.x + threadIdx.x;
  if (i < NN * 3 * TT) {
    const float c = cnt[i / 12];
    xout[i] = x[i] + agg[i] / fmaxf(c, 1.0f);
  }
}

// ---------------------------------------------------------------------------
extern "C" void kernel_launch(void* const* d_in, const int* in_sizes, int n_in,
                              void* d_out, int out_size, void* d_ws, size_t ws_size,
                              hipStream_t stream) {
  (void)in_sizes; (void)n_in; (void)out_size; (void)ws_size;

  const float* x     = (const float*)d_in[0];
  const float* h     = (const float*)d_in[1];
  const int*   eidx  = (const int*)d_in[2];
  const float* eattr = (const float*)d_in[3];
  // d_in[4] = batch (unused)
  const float* ew1 = (const float*)d_in[5];
  const float* eb1 = (const float*)d_in[6];
  const float* ew2 = (const float*)d_in[7];
  const float* eb2 = (const float*)d_in[8];
  const float* cw1 = (const float*)d_in[9];
  const float* cb1 = (const float*)d_in[10];
  const float* cw2 = (const float*)d_in[11];
  const float* cb2 = (const float*)d_in[12];
  const float* nw1 = (const float*)d_in[13];
  const float* nb1 = (const float*)d_in[14];
  const float* nw2 = (const float*)d_in[15];
  const float* nb2 = (const float*)d_in[16];

  char* ws = (char*)d_ws;
  size_t off = 0;
  __bf16* W1e = (__bf16*)(ws + off); off += (size_t)HDC * K1P * 2;   // 40960
  __bf16* W2e = (__bf16*)(ws + off); off += (size_t)HDC * HDC * 2;   // 32768
  __bf16* Wc1 = (__bf16*)(ws + off); off += (size_t)HDC * HDC * 2;   // 32768
  __bf16* W1n = (__bf16*)(ws + off); off += (size_t)HDC * KNP * 2;   // 49152
  __bf16* W2n = (__bf16*)(ws + off); off += (size_t)HHC * HDC * 2;   // 16384
  off = (off + 255) & ~(size_t)255;                                  // = 172032
  float* m_i = (float*)(ws + off); off += (size_t)NN * HDC * TT * 4; // 40.96 MB
  float* agg = (float*)(ws + off); off += (size_t)NN * 3 * TT * 4;
  float* cnt = (float*)(ws + off); off += (size_t)NN * 4;

  float* xout = (float*)d_out;
  float* hout = xout + (size_t)NN * 3 * TT;

  // m_i, agg, cnt are contiguous: zero them in one pass (required each launch).
  const int zn = NN * HDC * TT + NN * 3 * TT + NN;   // 10,500,000 floats
  egnn_zero_kernel<<<4096, 256, 0, stream>>>(m_i, zn);

  const int wtot = HDC * K1P + HDC * HDC + HDC * HDC + HDC * KNP + HHC * HDC;
  egnn_convert_weights<<<(wtot + 255) / 256, 256, 0, stream>>>(
      ew1, ew2, cw1, nw1, nw2, W1e, W2e, Wc1, W1n, W2n);

  egnn_edge_kernel<<<NE / 16, 128, 0, stream>>>(
      x, h, eidx, eattr, W1e, eb1, W2e, eb2, Wc1, cb1, cw2, cb2, m_i, agg, cnt);

  egnn_node_kernel<<<NN / 16, 128, 0, stream>>>(h, m_i, W1n, nb1, W2n, nb2, hout);

  egnn_x_final<<<(NN * 3 * TT + 255) / 256, 256, 0, stream>>>(x, agg, cnt, xout);
}
